// BertCRF_41300405518966
// MI455X (gfx1250) — matrix operand order repaired and checked by reading, not compiled, not run on hardware
//
#include <hip/hip_runtime.h>
#include <hip/hip_bf16.h>
#include <math.h>
#include <stdint.h>

typedef __attribute__((ext_vector_type(16))) _Float16 v16h;
typedef __attribute__((ext_vector_type(8)))  float    v8f;

static constexpr int Lc = 12, Hc = 768, NHc = 12, DHc = 64, FFc = 3072;
static constexpr int Kc = 9, Bc = 32, Sc = 256;
static constexpr int TOK = Bc * Sc;   // 8192 tokens

// ---------------------------------------------------------------------------
// Tensor Data Mover support (guarded; falls back to manual staging)
// ---------------------------------------------------------------------------
#if defined(__HIP_DEVICE_COMPILE__) && defined(__gfx1250__) && \
    __has_builtin(__builtin_amdgcn_tensor_load_to_lds) &&      \
    __has_builtin(__builtin_amdgcn_s_wait_tensorcnt)
#define HAVE_TDM 1
#endif

#ifdef HAVE_TDM
typedef __attribute__((ext_vector_type(4))) unsigned int v4u;
typedef __attribute__((ext_vector_type(8))) int          v8i;
typedef __attribute__((ext_vector_type(4))) int          v4i;

// 2D tile load: `rows` rows of 32 f16 elements, row stride lda (elements),
// into LDS rows of 40 halves (32 data + 8 pad). pad_interval=16 DW (code 3),
// pad_amount=4 DW (code 3), data_size=2B (code 1).
__device__ __forceinline__ void tdm_load_tile32(const void* gptr, unsigned lds_off,
                                                int rows, int lda_elems) {
    const unsigned long long ga = (unsigned long long)(uintptr_t)gptr;
    v4u g0;
    g0[0] = 1u;                                           // count=1, user mode
    g0[1] = lds_off;                                      // lds_addr
    g0[2] = (unsigned)(ga & 0xFFFFFFFFu);                 // global_addr[31:0]
    g0[3] = (unsigned)((ga >> 32) & 0x01FFFFFFu)          // global_addr[56:32]
          | (2u << 30);                                   // type = 2 ("image")
    v8i g1;
    g1[0] = (int)((1u << 16)      // data_size = 2 bytes
                | (1u << 20)      // pad_enable
                | (3u << 22)      // pad_interval: 16 DWORDs
                | (3u << 25));    // pad_amount:   4 DWORDs
    g1[1] = (int)(((unsigned)lda_elems & 0xFFFFu) << 16);            // tensor_dim0[15:0]
    g1[2] = (int)((((unsigned)lda_elems >> 16) & 0xFFFFu)
                | (((unsigned)rows & 0xFFFFu) << 16));               // dim0 hi | dim1 lo
    g1[3] = (int)((((unsigned)rows >> 16) & 0xFFFFu) | (32u << 16)); // dim1 hi | tile_dim0=32
    g1[4] = (int)((unsigned)rows & 0xFFFFu);                         // tile_dim1 = rows
    g1[5] = (int)(unsigned)lda_elems;                                // dim0_stride[31:0]
    g1[6] = 0;
    g1[7] = 0;
    const v4i zz4 = {0, 0, 0, 0};
    const v8i zz8 = {0, 0, 0, 0, 0, 0, 0, 0};
    __builtin_amdgcn_tensor_load_to_lds(g0, g1, zz4, zz4, zz8, 0);
}
#endif

// ---------------------------------------------------------------------------
// f32 -> f16 cast (weights)
// ---------------------------------------------------------------------------
__global__ void cast_f32_f16(const float* __restrict__ in,
                             _Float16* __restrict__ out, long long n) {
    long long i = (long long)blockIdx.x * blockDim.x + threadIdx.x;
    if (i < n) out[i] = (_Float16)in[i];
}

// ---------------------------------------------------------------------------
// WMMA helpers: CDNA5 fragment gather from LDS
//   A fragment 16x32 f16: lanes 0-15 hold K {0..7,16..23}; lanes 16-31: +8
//   B fragment 32x16 f16 (stored [n][k]): lanes 0-15 K {0..7,16..23}; lanes 16-31: +8
// ---------------------------------------------------------------------------
__device__ __forceinline__ v16h frag_a(const _Float16* rowp, int g) {
    v16h a;
#pragma unroll
    for (int j = 0; j < 8; ++j) {
        const int k0 = ((j < 4) ? (2 * j) : (16 + 2 * (j - 4))) + 8 * g;
        a[2 * j]     = rowp[k0];
        a[2 * j + 1] = rowp[k0 + 1];
    }
    return a;
}
__device__ __forceinline__ v16h frag_b(const _Float16* rowp, int g) {
    v16h b;
#pragma unroll
    for (int j = 0; j < 8; ++j) {
        const int kb = ((j < 4) ? (2 * j) : (16 + 2 * (j - 4))) + 8 * g;
        b[2 * j]     = rowp[kb];
        b[2 * j + 1] = rowp[kb + 1];
    }
    return b;
}

// ---------------------------------------------------------------------------
// Large-tile WMMA GEMM: 256 threads = 8 waves, tile 128(M) x 128(N), K-step 32.
// Wave w: rows (w&3)*32..+31, cols (w>>2)*64..+63  => 8 WMMA per K-step/wave.
// Batched/strided. M%128==0, N%128==0, Kd%32==0.
// ---------------------------------------------------------------------------
template <bool TRANSB>
__global__ void __launch_bounds__(256)
gemm_wmma_f16_128(const _Float16* __restrict__ A, long long sAb, long long sAh, int lda,
                  const _Float16* __restrict__ B, long long sBb, long long sBh, int ldb,
                  const float* __restrict__ bias,
                  float* __restrict__ C, _Float16* __restrict__ Ch,
                  long long sCb, long long sCh, int ldc,
                  int M, int N, int Kd, int zdiv) {
    __shared__ _Float16 As[128][40];   // [m][k] 32 data + 8 pad halves
    __shared__ _Float16 Bs[128][40];   // [n][k]

    const int t    = threadIdx.x;
    const int lane = t & 31;
    const int w    = t >> 5;
    const int wm2  = w & 3;            // 32-row slice
    const int wn2  = w >> 2;           // 64-col half
    const int r    = lane & 15;
    const int g    = lane >> 4;
    const int col0 = blockIdx.x * 128;
    const int row0 = blockIdx.y * 128;
    const int z    = blockIdx.z;
    const int zb   = z / zdiv;
    const int zh   = z % zdiv;

    const _Float16* Ab = A + zb * sAb + zh * sAh;
    const _Float16* Bb = B + zb * sBb + zh * sBh;

    const v8f vzero = {0.f, 0.f, 0.f, 0.f, 0.f, 0.f, 0.f, 0.f};
    v8f acc[2][4];
#pragma unroll
    for (int s = 0; s < 2; ++s)
#pragma unroll
        for (int wn = 0; wn < 4; ++wn) acc[s][wn] = vzero;

    for (int kk = 0; kk < Kd; kk += 32) {
#ifdef HAVE_TDM
        // ---- TDM: wave 0 DMAs the A tile (and transB B tile) into LDS ----
        if (w == 0) {
            tdm_load_tile32(Ab + (long long)row0 * lda + kk,
                            (unsigned)(uintptr_t)(void*)&As[0][0], 128, lda);
            if (TRANSB) {
                tdm_load_tile32(Bb + (long long)col0 * ldb + kk,
                                (unsigned)(uintptr_t)(void*)&Bs[0][0], 128, ldb);
            }
            __builtin_amdgcn_s_wait_tensorcnt(0);
        }
#else
        {   // ---- manual stage A tile 128x32 (16 halves / thread) ----
            const int arow = t >> 1;
            const int ach  = t & 1;
            const _Float16* gsrc = Ab + (long long)(row0 + arow) * lda + kk + ach * 16;
            const uint4* gp = reinterpret_cast<const uint4*>(gsrc);
            uint4 d0 = gp[0];
            uint4 d1 = gp[1];
            if (kk + 32 < Kd) __builtin_prefetch((const void*)(gsrc + 32), 0, 1);
            uint4* sp = reinterpret_cast<uint4*>(&As[arow][ach * 16]);
            sp[0] = d0; sp[1] = d1;
        }
        if (TRANSB) {
            const int nrow = t >> 1;
            const int nch  = t & 1;
            const _Float16* gsrc = Bb + (long long)(col0 + nrow) * ldb + kk + nch * 16;
            const uint4* gp = reinterpret_cast<const uint4*>(gsrc);
            uint4 d0 = gp[0];
            uint4 d1 = gp[1];
            uint4* sp = reinterpret_cast<uint4*>(&Bs[nrow][nch * 16]);
            sp[0] = d0; sp[1] = d1;
        }
#endif
        if (!TRANSB) {
            // ---- manual stage B 32x128 with transpose into Bs[n][k] ----
            const int krow = t >> 3;
            const int nch  = t & 7;
            const _Float16* gsrc = Bb + (long long)(kk + krow) * ldb + col0 + nch * 16;
            const uint4* gp = reinterpret_cast<const uint4*>(gsrc);
            uint4 d0 = gp[0];
            uint4 d1 = gp[1];
            if (kk + 32 < Kd) __builtin_prefetch((const void*)(gsrc + 32ll * ldb), 0, 1);
            _Float16 tmp[16];
            *reinterpret_cast<uint4*>(&tmp[0]) = d0;
            *reinterpret_cast<uint4*>(&tmp[8]) = d1;
#pragma unroll
            for (int i = 0; i < 16; ++i) Bs[nch * 16 + i][krow] = tmp[i];
        }
        __syncthreads();

        v16h af0 = frag_a(&As[wm2 * 32 + r][0], g);
        v16h af1 = frag_a(&As[wm2 * 32 + 16 + r][0], g);
#pragma unroll
        for (int wn = 0; wn < 4; ++wn) {
            v16h bf = frag_b(&Bs[wn2 * 64 + wn * 16 + r][0], g);
            acc[0][wn] = __builtin_amdgcn_wmma_f32_16x16x32_f16(
                false, af0, false, bf, (short)0, acc[0][wn], false, false);
            acc[1][wn] = __builtin_amdgcn_wmma_f32_16x16x32_f16(
                false, af1, false, bf, (short)0, acc[1][wn], false, false);
        }
        __syncthreads();
    }

    const long long zoff = zb * sCb + zh * sCh;
#pragma unroll
    for (int wn = 0; wn < 4; ++wn) {
        const int col = col0 + wn2 * 64 + wn * 16 + r;
        const float bv = bias ? bias[col] : 0.0f;
#pragma unroll
        for (int s = 0; s < 2; ++s) {
#pragma unroll
            for (int i = 0; i < 8; ++i) {
                const int row = row0 + wm2 * 32 + s * 16 + g * 8 + i;
                const float val = acc[s][wn][i] + bv;
                const long long off = zoff + (long long)row * ldc + col;
                if (C)  C[off]  = val;
                if (Ch) Ch[off] = (_Float16)val;
            }
        }
    }
}

// ---------------------------------------------------------------------------
// Small-tile WMMA GEMM (64x64, 128 threads = 4 waves) — used where N==64.
// ---------------------------------------------------------------------------
__global__ void __launch_bounds__(128)
gemm_wmma_f16_64(const _Float16* __restrict__ A, long long sAb, long long sAh, int lda,
                 const _Float16* __restrict__ B, long long sBb, long long sBh, int ldb,
                 const float* __restrict__ bias,
                 float* __restrict__ C, _Float16* __restrict__ Ch,
                 long long sCb, long long sCh, int ldc,
                 int M, int N, int Kd, int zdiv) {
    __shared__ _Float16 As[64][40];
    __shared__ _Float16 Bs[64][40];

    const int t    = threadIdx.x;
    const int lane = t & 31;
    const int wm   = t >> 5;
    const int r    = lane & 15;
    const int g    = lane >> 4;
    const int col0 = blockIdx.x * 64;
    const int row0 = blockIdx.y * 64;
    const int z    = blockIdx.z;
    const int zb   = z / zdiv;
    const int zh   = z % zdiv;

    const _Float16* Ab = A + zb * sAb + zh * sAh;
    const _Float16* Bb = B + zb * sBb + zh * sBh;

    const v8f vzero = {0.f, 0.f, 0.f, 0.f, 0.f, 0.f, 0.f, 0.f};
    v8f acc[4];
    acc[0] = vzero; acc[1] = vzero; acc[2] = vzero; acc[3] = vzero;

    for (int kk = 0; kk < Kd; kk += 32) {
        {
            const int arow = t >> 1;
            const int ach  = t & 1;
            const uint4* gp = reinterpret_cast<const uint4*>(
                Ab + (long long)(row0 + arow) * lda + kk + ach * 16);
            uint4 d0 = gp[0];
            uint4 d1 = gp[1];
            uint4* sp = reinterpret_cast<uint4*>(&As[arow][ach * 16]);
            sp[0] = d0; sp[1] = d1;
        }
        {   // non-transposed B: stage 32x64 with transpose into Bs[n][k]
            const int krow = t >> 2;
            const int nch  = t & 3;
            const uint4* gp = reinterpret_cast<const uint4*>(
                Bb + (long long)(kk + krow) * ldb + col0 + nch * 16);
            uint4 d0 = gp[0];
            uint4 d1 = gp[1];
            _Float16 tmp[16];
            *reinterpret_cast<uint4*>(&tmp[0]) = d0;
            *reinterpret_cast<uint4*>(&tmp[8]) = d1;
#pragma unroll
            for (int i = 0; i < 16; ++i) Bs[nch * 16 + i][krow] = tmp[i];
        }
        __syncthreads();

        v16h af = frag_a(&As[wm * 16 + r][0], g);
#pragma unroll
        for (int wn = 0; wn < 4; ++wn) {
            v16h bf = frag_b(&Bs[wn * 16 + r][0], g);
            acc[wn] = __builtin_amdgcn_wmma_f32_16x16x32_f16(
                false, af, false, bf, (short)0, acc[wn], false, false);
        }
        __syncthreads();
    }

    const long long zoff = zb * sCb + zh * sCh;
#pragma unroll
    for (int wn = 0; wn < 4; ++wn) {
        const int col = col0 + wn * 16 + r;
        const float bv = bias ? bias[col] : 0.0f;
#pragma unroll
        for (int i = 0; i < 8; ++i) {
            const int row = row0 + wm * 16 + g * 8 + i;
            const float val = acc[wn][i] + bv;
            const long long off = zoff + (long long)row * ldc + col;
            if (C)  C[off]  = val;
            if (Ch) Ch[off] = (_Float16)val;
        }
    }
}

// ---------------------------------------------------------------------------
// Embedding lookup + LayerNorm (writes f32 x and f16 x_h)
// ---------------------------------------------------------------------------
__global__ void __launch_bounds__(256)
embed_ln(const int* __restrict__ ids, const float* __restrict__ wemb,
         const float* __restrict__ pemb, const float* __restrict__ temb,
         const float* __restrict__ gg, const float* __restrict__ bb,
         float* __restrict__ x, _Float16* __restrict__ xh) {
    const int row = blockIdx.x;
    const int s   = row % Sc;
    __shared__ float buf[Hc];
    __shared__ float red[256];
    const int t  = threadIdx.x;
    const long long id = ids[row];

    float lsum = 0.f;
    for (int c = t; c < Hc; c += 256) {
        float v = wemb[id * Hc + c] + pemb[(long long)s * Hc + c] + temb[c];
        buf[c] = v;
        lsum += v;
    }
    red[t] = lsum; __syncthreads();
    for (int st = 128; st > 0; st >>= 1) {
        if (t < st) red[t] += red[t + st];
        __syncthreads();
    }
    const float mean = red[0] / Hc;
    __syncthreads();
    float lss = 0.f;
    for (int c = t; c < Hc; c += 256) {
        float d = buf[c] - mean;
        lss += d * d;
    }
    red[t] = lss; __syncthreads();
    for (int st = 128; st > 0; st >>= 1) {
        if (t < st) red[t] += red[t + st];
        __syncthreads();
    }
    const float inv = rsqrtf(red[0] / Hc + 1e-12f);
    const long long base = (long long)row * Hc;
    for (int c = t; c < Hc; c += 256) {
        float o = (buf[c] - mean) * inv * gg[c] + bb[c];
        x[base + c]  = o;
        xh[base + c] = (_Float16)o;
    }
}

// ---------------------------------------------------------------------------
// x = LayerNorm(x + y) ; writes f32 x (in place) and f16 x_h
// ---------------------------------------------------------------------------
__global__ void __launch_bounds__(256)
add_ln(float* x, const float* __restrict__ y,
       const float* __restrict__ gg, const float* __restrict__ bb,
       _Float16* __restrict__ xh) {
    const int row = blockIdx.x;
    const long long base = (long long)row * Hc;
    __shared__ float buf[Hc];
    __shared__ float red[256];
    const int t = threadIdx.x;

    float lsum = 0.f;
    for (int c = t; c < Hc; c += 256) {
        float v = x[base + c] + y[base + c];
        buf[c] = v;
        lsum += v;
    }
    red[t] = lsum; __syncthreads();
    for (int st = 128; st > 0; st >>= 1) {
        if (t < st) red[t] += red[t + st];
        __syncthreads();
    }
    const float mean = red[0] / Hc;
    __syncthreads();
    float lss = 0.f;
    for (int c = t; c < Hc; c += 256) {
        float d = buf[c] - mean;
        lss += d * d;
    }
    red[t] = lss; __syncthreads();
    for (int st = 128; st > 0; st >>= 1) {
        if (t < st) red[t] += red[t + st];
        __syncthreads();
    }
    const float inv = rsqrtf(red[0] / Hc + 1e-12f);
    for (int c = t; c < Hc; c += 256) {
        float o = (buf[c] - mean) * inv * gg[c] + bb[c];
        x[base + c]  = o;
        xh[base + c] = (_Float16)o;
    }
}

// ---------------------------------------------------------------------------
// Softmax over attention scores with mask bias and 1/sqrt(DH) scale.
// ---------------------------------------------------------------------------
__global__ void __launch_bounds__(256)
softmax_mask(const float* __restrict__ sc, const int* __restrict__ mask,
             _Float16* __restrict__ ph) {
    const long long idx = blockIdx.x;            // (b*NH + h)*S + q
    const int b = (int)(idx / ((long long)NHc * Sc));
    const int t = threadIdx.x;
    __shared__ float red[256];

    float v = sc[idx * Sc + t] * 0.125f
            + (1.0f - (float)mask[b * Sc + t]) * -10000.0f;

    red[t] = v; __syncthreads();
    for (int st = 128; st > 0; st >>= 1) {
        if (t < st) red[t] = fmaxf(red[t], red[t + st]);
        __syncthreads();
    }
    const float mx = red[0];
    __syncthreads();
    const float e = expf(v - mx);
    red[t] = e; __syncthreads();
    for (int st = 128; st > 0; st >>= 1) {
        if (t < st) red[t] += red[t + st];
        __syncthreads();
    }
    ph[idx * Sc + t] = (_Float16)(e / red[0]);
}

// ---------------------------------------------------------------------------
// Exact GELU (erf-based) + f16 cast
// ---------------------------------------------------------------------------
__global__ void gelu_cast(const float* __restrict__ in,
                          _Float16* __restrict__ out, long long n) {
    long long i = (long long)blockIdx.x * blockDim.x + threadIdx.x;
    if (i < n) {
        float v = in[i];
        out[i] = (_Float16)(0.5f * v * (1.0f + erff(v * 0.70710678118654752f)));
    }
}

// ---------------------------------------------------------------------------
// Classifier head: emissions[row][k] = x[row] . cls_W[:,k] + cls_b[k]
// ---------------------------------------------------------------------------
__global__ void classifier(const float* __restrict__ x, const float* __restrict__ W,
                           const float* __restrict__ b, float* __restrict__ em) {
    const int i = blockIdx.x * blockDim.x + threadIdx.x;
    if (i >= TOK * Kc) return;
    const int row = i / Kc;
    const int k   = i % Kc;
    const float* xr = x + (long long)row * Hc;
    float acc = b[k];
    for (int j = 0; j < Hc; ++j) acc += xr[j] * W[j * Kc + k];
    em[i] = acc;
}

// ---------------------------------------------------------------------------
// CRF forward: one wave per batch row. alpha recursion, logsumexp over K=9.
// ---------------------------------------------------------------------------
__global__ void __launch_bounds__(32)
crf_fwd(const float* __restrict__ em, const int* __restrict__ labels,
        const int* __restrict__ mask, const float* __restrict__ cs,
        const float* __restrict__ ce, const float* __restrict__ tr,
        float* __restrict__ res) {
    const int b = blockIdx.x;
    const int j = threadIdx.x;
    __shared__ float alpha[Kc];
    __shared__ float trs[Kc * Kc];
    for (int i = j; i < Kc * Kc; i += 32) trs[i] = tr[i];
    const float* eb = em + (long long)b * Sc * Kc;
    if (j < Kc) alpha[j] = cs[j] + eb[j];
    __syncthreads();

    for (int t = 1; t < Sc; ++t) {
        float nxt = 0.f;
        if (j < Kc) {
            float mx = -1e30f;
            for (int i = 0; i < Kc; ++i)
                mx = fmaxf(mx, alpha[i] + trs[i * Kc + j]);
            float s = 0.f;
            for (int i = 0; i < Kc; ++i)
                s += expf(alpha[i] + trs[i * Kc + j] - mx);
            nxt = mx + logf(s) + eb[t * Kc + j];
        }
        __syncthreads();
        if (j < Kc && mask[b * Sc + t]) alpha[j] = nxt;
        __syncthreads();
    }

    if (j == 0) {
        float mx = -1e30f;
        for (int i = 0; i < Kc; ++i) mx = fmaxf(mx, alpha[i] + ce[i]);
        float s = 0.f;
        for (int i = 0; i < Kc; ++i) s += expf(alpha[i] + ce[i] - mx);
        const float denom = mx + logf(s);

        const int* lb = labels + b * Sc;
        const int* mk = mask + b * Sc;
        float num = cs[lb[0]] + eb[lb[0]];
        int cnt = mk[0];
        for (int t = 1; t < Sc; ++t) {
            if (mk[t]) num += trs[lb[t - 1] * Kc + lb[t]] + eb[t * Kc + lb[t]];
            cnt += mk[t];
        }
        num += ce[lb[cnt - 1]];
        res[b] = num - denom;
    }
}

__global__ void finalize(const float* __restrict__ res, float* __restrict__ out) {
    if (threadIdx.x == 0) {
        float s = 0.f;
        for (int b = 0; b < Bc; ++b) s += res[b];
        out[0] = -s;
    }
}

// ---------------------------------------------------------------------------
// Host driver
// ---------------------------------------------------------------------------
extern "C" void kernel_launch(void* const* d_in, const int* in_sizes, int n_in,
                              void* d_out, int out_size, void* d_ws, size_t ws_size,
                              hipStream_t stream) {
    (void)in_sizes; (void)n_in; (void)out_size; (void)ws_size;

    const int*   ids    = (const int*)d_in[0];
    const int*   amask  = (const int*)d_in[1];
    const int*   labels = (const int*)d_in[2];
    const float* wemb   = (const float*)d_in[3];
    const float* pemb   = (const float*)d_in[4];
    const float* temb   = (const float*)d_in[5];
    const float* elng   = (const float*)d_in[6];
    const float* elnb   = (const float*)d_in[7];
    const float* Wq = (const float*)d_in[8];  const float* bq = (const float*)d_in[9];
    const float* Wk = (const float*)d_in[10]; const float* bk = (const float*)d_in[11];
    const float* Wv = (const float*)d_in[12]; const float* bv = (const float*)d_in[13];
    const float* Wo = (const float*)d_in[14]; const float* bo = (const float*)d_in[15];
    const float* ln1g = (const float*)d_in[16]; const float* ln1b = (const float*)d_in[17];
    const float* W1 = (const float*)d_in[18]; const float* b1 = (const float*)d_in[19];
    const float* W2 = (const float*)d_in[20]; const float* b2 = (const float*)d_in[21];
    const float* ln2g = (const float*)d_in[22]; const float* ln2b = (const float*)d_in[23];
    const float* clsW = (const float*)d_in[24]; const float* clsb = (const float*)d_in[25];
    const float* cstart = (const float*)d_in[26];
    const float* cend   = (const float*)d_in[27];
    const float* ctrans = (const float*)d_in[28];

    char* p = (char*)d_ws;
    auto alloc = [&](size_t bytes) -> void* {
        void* r = (void*)p;
        p += (bytes + 255) & ~(size_t)255;
        return r;
    };

    const long long nQKV = (long long)Lc * Hc * Hc;
    const long long nFF  = (long long)Lc * Hc * FFc;

    _Float16* wq_h = (_Float16*)alloc(nQKV * 2);
    _Float16* wk_h = (_Float16*)alloc(nQKV * 2);
    _Float16* wv_h = (_Float16*)alloc(nQKV * 2);
    _Float16* wo_h = (_Float16*)alloc(nQKV * 2);
    _Float16* w1_h = (_Float16*)alloc(nFF * 2);
    _Float16* w2_h = (_Float16*)alloc(nFF * 2);

    float*    x    = (float*)   alloc((long long)TOK * Hc * 4);
    _Float16* x_h  = (_Float16*)alloc((long long)TOK * Hc * 2);
    _Float16* q_h  = (_Float16*)alloc((long long)TOK * Hc * 2);
    _Float16* k_h  = (_Float16*)alloc((long long)TOK * Hc * 2);
    _Float16* v_h  = (_Float16*)alloc((long long)TOK * Hc * 2);
    float*    scores  = (float*)   alloc((long long)Bc * NHc * Sc * Sc * 4);
    _Float16* probs_h = (_Float16*)alloc((long long)Bc * NHc * Sc * Sc * 2);
    _Float16* ctx_h   = (_Float16*)alloc((long long)TOK * Hc * 2);
    float*    yout    = (float*)   alloc((long long)TOK * Hc * 4);
    float*    emis    = (float*)   alloc((long long)TOK * Kc * 4);
    float*    res     = (float*)   alloc(Bc * 4);

    float*    h1   = scores;    // alias: equal sizes, disjoint lifetimes
    _Float16* h1_h = probs_h;   // alias: equal sizes, disjoint lifetimes

    {
        const int TB = 256;
        cast_f32_f16<<<(unsigned)((nQKV + TB - 1) / TB), TB, 0, stream>>>(Wq, wq_h, nQKV);
        cast_f32_f16<<<(unsigned)((nQKV + TB - 1) / TB), TB, 0, stream>>>(Wk, wk_h, nQKV);
        cast_f32_f16<<<(unsigned)((nQKV + TB - 1) / TB), TB, 0, stream>>>(Wv, wv_h, nQKV);
        cast_f32_f16<<<(unsigned)((nQKV + TB - 1) / TB), TB, 0, stream>>>(Wo, wo_h, nQKV);
        cast_f32_f16<<<(unsigned)((nFF  + TB - 1) / TB), TB, 0, stream>>>(W1, w1_h, nFF);
        cast_f32_f16<<<(unsigned)((nFF  + TB - 1) / TB), TB, 0, stream>>>(W2, w2_h, nFF);
    }

    // Large-tile GEMM (M%128==0, N%128==0)
    auto gemmBig = [&](const _Float16* A, long long sAb, long long sAh, int lda,
                       const _Float16* B, long long sBb, long long sBh, int ldb,
                       bool transB, const float* bias,
                       float* C, _Float16* Ch, long long sCb, long long sCh, int ldc,
                       int M, int N, int Kd, int batches, int zdiv) {
        dim3 grid((unsigned)(N / 128), (unsigned)(M / 128), (unsigned)batches);
        if (transB)
            gemm_wmma_f16_128<true><<<grid, dim3(256), 0, stream>>>(
                A, sAb, sAh, lda, B, sBb, sBh, ldb, bias, C, Ch, sCb, sCh, ldc,
                M, N, Kd, zdiv);
        else
            gemm_wmma_f16_128<false><<<grid, dim3(256), 0, stream>>>(
                A, sAb, sAh, lda, B, sBb, sBh, ldb, bias, C, Ch, sCb, sCh, ldc,
                M, N, Kd, zdiv);
    };
    // Small-tile GEMM (non-trans B), for N==64 cases
    auto gemmSmall = [&](const _Float16* A, long long sAb, long long sAh, int lda,
                         const _Float16* B, long long sBb, long long sBh, int ldb,
                         const float* bias,
                         float* C, _Float16* Ch, long long sCb, long long sCh, int ldc,
                         int M, int N, int Kd, int batches, int zdiv) {
        dim3 grid((unsigned)(N / 64), (unsigned)(M / 64), (unsigned)batches);
        gemm_wmma_f16_64<<<grid, dim3(128), 0, stream>>>(
            A, sAb, sAh, lda, B, sBb, sBh, ldb, bias, C, Ch, sCb, sCh, ldc,
            M, N, Kd, zdiv);
    };

    embed_ln<<<TOK, 256, 0, stream>>>(ids, wemb, pemb, temb, elng, elnb, x, x_h);

    const long long sBSH = (long long)Sc * Hc;
    const long long sHSS = (long long)Sc * Sc;

    for (int l = 0; l < Lc; ++l) {
        const _Float16* wqh = wq_h + (long long)l * Hc * Hc;
        const _Float16* wkh = wk_h + (long long)l * Hc * Hc;
        const _Float16* wvh = wv_h + (long long)l * Hc * Hc;
        const _Float16* woh = wo_h + (long long)l * Hc * Hc;
        const _Float16* w1h = w1_h + (long long)l * Hc * FFc;
        const _Float16* w2h = w2_h + (long long)l * FFc * Hc;

        // Q/K/V projections (f16 out only)
        gemmBig(x_h, 0, 0, Hc, wqh, 0, 0, Hc, false, bq + l * Hc,
                nullptr, q_h, 0, 0, Hc, TOK, Hc, Hc, 1, 1);
        gemmBig(x_h, 0, 0, Hc, wkh, 0, 0, Hc, false, bk + l * Hc,
                nullptr, k_h, 0, 0, Hc, TOK, Hc, Hc, 1, 1);
        gemmBig(x_h, 0, 0, Hc, wvh, 0, 0, Hc, false, bv + l * Hc,
                nullptr, v_h, 0, 0, Hc, TOK, Hc, Hc, 1, 1);

        // scores[b,h] = q[b,:,h,:] @ k[b,:,h,:]^T
        gemmBig(q_h, sBSH, DHc, Hc, k_h, sBSH, DHc, Hc, true, nullptr,
                scores, nullptr, (long long)NHc * sHSS, sHSS, Sc,
                Sc, Sc, DHc, Bc * NHc, NHc);

        softmax_mask<<<Bc * NHc * Sc, 256, 0, stream>>>(scores, amask, probs_h);

        // ctx[b,:,h,:] = probs[b,h] @ v[b,:,h,:]   (N=64 -> small kernel)
        gemmSmall(probs_h, (long long)NHc * sHSS, sHSS, Sc, v_h, sBSH, DHc, Hc,
                  nullptr, nullptr, ctx_h, sBSH, DHc, Hc,
                  Sc, DHc, Sc, Bc * NHc, NHc);

        // output projection
        gemmBig(ctx_h, 0, 0, Hc, woh, 0, 0, Hc, false, bo + l * Hc,
                yout, nullptr, 0, 0, Hc, TOK, Hc, Hc, 1, 1);

        add_ln<<<TOK, 256, 0, stream>>>(x, yout, ln1g + l * Hc, ln1b + l * Hc, x_h);

        // FFN
        gemmBig(x_h, 0, 0, Hc, w1h, 0, 0, FFc, false, b1 + l * FFc,
                h1, nullptr, 0, 0, FFc, TOK, FFc, Hc, 1, 1);
        gelu_cast<<<(unsigned)(((long long)TOK * FFc + 255) / 256), 256, 0, stream>>>(
            h1, h1_h, (long long)TOK * FFc);
        gemmBig(h1_h, 0, 0, FFc, w2h, 0, 0, Hc, false, b2 + l * Hc,
                yout, nullptr, 0, 0, Hc, TOK, Hc, FFc, 1, 1);

        add_ln<<<TOK, 256, 0, stream>>>(x, yout, ln2g + l * Hc, ln2b + l * Hc, x_h);
    }

    classifier<<<(TOK * Kc + 255) / 256, 256, 0, stream>>>(x, clsW, clsb, emis);
    crf_fwd<<<Bc, 32, 0, stream>>>(emis, labels, amask, cstart, cend, ctrans, res);
    finalize<<<1, 32, 0, stream>>>(res, (float*)d_out);
}
